// Parser_75514114998803
// MI455X (gfx1250) — matrix operand order repaired and checked by reading, not compiled, not run on hardware
//
#include <hip/hip_runtime.h>
#include <hip/hip_bf16.h>

#define B_   32
#define L_   512
#define T_   256
#define D_   768
#define H_   100
#define G_   400   // 4*H
#define O_   50
#define NTOK 8192  // B_*T_
#define WSTR 104   // W_hh LDS row stride (16B aligned, conflict-benign)

typedef __attribute__((ext_vector_type(16))) _Float16 v16h;
typedef __attribute__((ext_vector_type(8)))  float    v8f;

// ---------------- WMMA helpers (CDNA5 16x16x32 f16 -> f32) ----------------
// A-frag: MxK=16x32 row-major source. lane m=lane&15, half=lane>>4,
// VGPR v holds K pair k0 = half*16 + 2v.
__device__ __forceinline__ v16h load_a_frag(const _Float16* base, int stride, int lane) {
    int m = lane & 15, half = lane >> 4;
    v16h a;
#pragma unroll
    for (int v = 0; v < 8; ++v) {
        int k0 = half * 16 + 2 * v;
        a[2 * v]     = base[m * stride + k0];
        a[2 * v + 1] = base[m * stride + k0 + 1];
    }
    return a;
}
// B-frag from row-major KxN source: lanes 0-15 hold K=0..15, lanes 16-31 K=16..31,
// VGPR v holds K pair k0 = half*16 + 2v; n = lane&15.
__device__ __forceinline__ v16h load_b_frag_kn(const _Float16* base, int stride, int lane) {
    int n = lane & 15, half = lane >> 4;
    v16h b;
#pragma unroll
    for (int v = 0; v < 8; ++v) {
        int k0 = half * 16 + 2 * v;
        b[2 * v]     = base[k0 * stride + n];
        b[2 * v + 1] = base[(k0 + 1) * stride + n];
    }
    return b;
}
// B-frag from row-major NxK source (e.g. weight matrices stored [n][k]).
__device__ __forceinline__ v16h load_b_frag_nk(const _Float16* base, int stride, int lane) {
    int n = lane & 15, half = lane >> 4;
    v16h b;
#pragma unroll
    for (int v = 0; v < 8; ++v) {
        int k0 = half * 16 + 2 * v;
        b[2 * v]     = base[n * stride + k0];
        b[2 * v + 1] = base[n * stride + k0 + 1];
    }
    return b;
}
__device__ __forceinline__ v8f wmma16(v16h a, v16h b, v8f c) {
    return __builtin_amdgcn_wmma_f32_16x16x32_f16(false, a, false, b, (short)0, c, false, false);
}
__device__ __forceinline__ v8f zero8() {
    v8f c = {};
    return c;
}

// ---------------- CDNA5 async global->LDS copy (ASYNCcnt-tracked) ----------------
// VFLAT async form: VDST = per-lane LDS byte offset, VADDR = 64-bit global address.
__device__ __forceinline__ void async_copy_b128(void* lds_dst, const void* gsrc) {
    unsigned lds_off = (unsigned)(uintptr_t)lds_dst;          // addrspace(3) offset
    unsigned long long ga = (unsigned long long)(uintptr_t)gsrc;
    asm volatile("global_load_async_to_lds_b128 %0, %1, off"
                 :: "v"(lds_off), "v"(ga)
                 : "memory");
}
__device__ __forceinline__ void async_wait0() {
    asm volatile("s_wait_asynccnt 0" ::: "memory");
}

// ---------------- Kernel 1: segment mean (bert_out, f16) ----------------
__device__ __forceinline__ int lower_bound_i32(const int* a, int n, int val) {
    int lo = 0, hi = n;
    while (lo < hi) { int mid = (lo + hi) >> 1; if (a[mid] < val) lo = mid + 1; else hi = mid; }
    return lo;
}

__global__ void seg_mean_kernel(const float* __restrict__ hiddens,
                                const int* __restrict__ b2t,
                                _Float16* __restrict__ bert_h) {
    int bt = blockIdx.x;            // b*T + t
    int b = bt >> 8, t = bt & 255;
    __shared__ int s_lo, s_hi;
    if (threadIdx.x == 0) {
        const int* row = b2t + b * L_;
        s_lo = lower_bound_i32(row, L_, t);
        s_hi = lower_bound_i32(row, L_, t + 1);
    }
    __syncthreads();
    int lo = s_lo, hi = s_hi;
    float inv = 1.0f / (3.0f * (float)(hi - lo));
    for (int d = threadIdx.x; d < D_; d += blockDim.x) {
        float acc = 0.f;
        for (int l = lo; l < hi; ++l) {
            size_t base = ((size_t)b * (L_ + 1) + (l + 1)) * D_ + d;
            // 151 MB read-once stream: non-temporal so it doesn't wash L2
            acc += __builtin_nontemporal_load(&hiddens[base])
                 + __builtin_nontemporal_load(&hiddens[base + (size_t)B_ * (L_ + 1) * D_])
                 + __builtin_nontemporal_load(&hiddens[base + (size_t)2 * B_ * (L_ + 1) * D_]);
        }
        bert_h[(size_t)bt * D_ + d] = (_Float16)(acc * inv);
    }
}

// ---------------- Kernel 2: convert W_ih (both dirs) to f16 ----------------
__global__ void convert_wih(const float* __restrict__ wf, const float* __restrict__ wb,
                            _Float16* __restrict__ out) {
    int i = blockIdx.x * blockDim.x + threadIdx.x;
    const int n = G_ * D_;
    if (i < n)            out[i] = (_Float16)wf[i];
    else if (i < 2 * n)   out[i] = (_Float16)wb[i - n];
}

// ---------------- Kernel 3: xg = X @ W_ih^T + b_ih + b_hh (WMMA) ----------------
// M=8192 tokens, N=400 gates, K=768, x2 directions. One 16x16 tile per wave.
__global__ void xg_gemm(const _Float16* __restrict__ x, const _Float16* __restrict__ wih,
                        const float* __restrict__ bihf, const float* __restrict__ bhhf,
                        const float* __restrict__ bihb, const float* __restrict__ bhhb,
                        float* __restrict__ xg) {
    int wave = blockIdx.x * (blockDim.x >> 5) + (threadIdx.x >> 5);
    int lane = threadIdx.x & 31;
    const int MT = NTOK / 16, NT = G_ / 16;           // 512, 25
    int dir = wave / (MT * NT);
    int rem = wave % (MT * NT);
    int mt = rem / NT, nt = rem % NT;
    const _Float16* A = x + (size_t)mt * 16 * D_;
    const _Float16* W = wih + (size_t)dir * G_ * D_ + (size_t)nt * 16 * D_;
    v8f c = zero8();
#pragma unroll 4
    for (int kt = 0; kt < D_ / 32; ++kt) {
        v16h a = load_a_frag(A + kt * 32, D_, lane);
        v16h b = load_b_frag_nk(W + kt * 32, D_, lane);
        c = wmma16(a, b, c);
    }
    int n = nt * 16 + (lane & 15);
    int half = lane >> 4;
    float bias = dir ? (bihb[n] + bhhb[n]) : (bihf[n] + bhhf[n]);
#pragma unroll
    for (int r = 0; r < 8; ++r) {
        int m = mt * 16 + r + half * 8;
        xg[((size_t)dir * NTOK + m) * G_ + n] = c[r] + bias;
    }
}

// ---------------- Kernel 4: LSTM recurrence (serial; W_hh LDS-resident) ----------
// One workgroup per (dir, batch). 64 independent recurrences run concurrently.
// W_hh kept in the WGP's 320 KB LDS; dots read float4 (ds_load_b128).
__global__ void lstm_rec(const float* __restrict__ whhf, const float* __restrict__ whhb,
                         const float* __restrict__ xg, float* __restrict__ h_all) {
    extern __shared__ float sm[];
    float* Whh  = sm;                 // 400 x WSTR
    float* gld  = sm + G_ * WSTR;     // 400
    float* hld  = gld + G_;           // 100 (+pad to 104)
    int dir = blockIdx.x >> 5;
    int b   = blockIdx.x & 31;
    const float* W = dir ? whhb : whhf;
    for (int i = threadIdx.x; i < G_ * H_; i += blockDim.x)
        Whh[(i / H_) * WSTR + (i % H_)] = W[i];
    for (int i = threadIdx.x; i < H_; i += blockDim.x) hld[i] = 0.f;
    __syncthreads();
    int tid = threadIdx.x;
    float cst = 0.f;
    for (int s = 0; s < T_; ++s) {
        int t = dir ? (T_ - 1 - s) : s;
        if (tid < G_) {
            float acc = xg[((size_t)dir * NTOK + b * T_ + t) * G_ + tid];
            const float4* wr = (const float4*)(Whh + tid * WSTR);
            const float4* hv = (const float4*)hld;
#pragma unroll
            for (int k4 = 0; k4 < H_ / 4; ++k4) {
                float4 w = wr[k4], h = hv[k4];
                acc += w.x * h.x + w.y * h.y + w.z * h.z + w.w * h.w;
            }
            gld[tid] = acc;
        }
        __syncthreads();
        if (tid < H_) {
            float gi = gld[tid], gf = gld[H_ + tid], gg = gld[2 * H_ + tid], go = gld[3 * H_ + tid];
            float si = 1.f / (1.f + __expf(-gi));
            float sf = 1.f / (1.f + __expf(-gf));
            float so = 1.f / (1.f + __expf(-go));
            cst = sf * cst + si * tanhf(gg);
            float h = so * tanhf(cst);
            hld[tid] = h;
            h_all[((size_t)dir * NTOK + b * T_ + t) * H_ + tid] = h;
        }
        __syncthreads();
    }
}

// ---------------- Kernel 5: MLP projections h1u/h2u/h1r/h2r (f16, padded 32) ----
__global__ void proj_kernel(const float* __restrict__ h_all,
                            const float* __restrict__ W1u, const float* __restrict__ b1u,
                            const float* __restrict__ W2u, const float* __restrict__ b2u,
                            const float* __restrict__ W1r, const float* __restrict__ b1r,
                            const float* __restrict__ W2r, const float* __restrict__ b2r,
                            _Float16* __restrict__ h1u, _Float16* __restrict__ h2u,
                            _Float16* __restrict__ h1r, _Float16* __restrict__ h2r) {
    __shared__ float xc[2 * H_];
    int tok = blockIdx.x;
    for (int i = threadIdx.x; i < 2 * H_; i += blockDim.x) {
        int d2 = (i < H_) ? 0 : 1;
        int k  = (i < H_) ? i : i - H_;
        xc[i] = h_all[((size_t)d2 * NTOK + tok) * H_ + k];
    }
    __syncthreads();
    int mat = threadIdx.x >> 5, col = threadIdx.x & 31;
    const float* Wm; const float* bm; _Float16* om;
    switch (mat) {
        case 0:  Wm = W1u; bm = b1u; om = h1u; break;
        case 1:  Wm = W2u; bm = b2u; om = h2u; break;
        case 2:  Wm = W1r; bm = b1r; om = h1r; break;
        default: Wm = W2r; bm = b2r; om = h2r; break;
    }
    float val;
    if (col < 30) {
        float acc = bm[col];
#pragma unroll 4
        for (int k = 0; k < 2 * H_; ++k) acc += xc[k] * Wm[col * (2 * H_) + k];
        val = fmaxf(acc, 0.f);
    } else {
        val = (col == 30) ? 1.0f : 0.0f;   // ones column; col 31 = zero pad
    }
    om[(size_t)tok * 32 + col] = (_Float16)val;
}

// ---------------- Kernel 6: unlabeled biaffine (O=1), WMMA K=32 --------------
// out_u[b,x,y] = (h1u[b,x] @ Uu) . h2u[b,y].   Per block: 64 x-rows, all 256 y.
__global__ void biaffine_u(const _Float16* __restrict__ h1u, const _Float16* __restrict__ h2u,
                           const float* __restrict__ Uu, const float* __restrict__ biasu,
                           float* __restrict__ out) {
    __shared__ __align__(16) _Float16 h2s[T_ * 32];   // 16 KB
    __shared__ __align__(16) _Float16 g1s[64 * 32];   // 4 KB
    int b = blockIdx.x >> 2, xbase = (blockIdx.x & 3) * 64;
    // async DMA of h2u[b] tile straight into LDS (no VGPR round-trip)
    for (int i = threadIdx.x; i < T_ * 32 / 8; i += blockDim.x)
        async_copy_b128(&h2s[i * 8], &h2u[(size_t)b * T_ * 32 + i * 8]);
    for (int e = threadIdx.x; e < 64 * 32; e += blockDim.x) {
        int xr = e >> 5, j = e & 31;
        float acc = 0.f;
        if (j < 31) {
            const _Float16* h1row = h1u + ((size_t)b * T_ + xbase + xr) * 32;
#pragma unroll 4
            for (int i2 = 0; i2 < 31; ++i2) acc += (float)h1row[i2] * Uu[i2 * 31 + j];
        }
        g1s[e] = (_Float16)acc;
    }
    async_wait0();
    __syncthreads();
    int wave = threadIdx.x >> 5, lane = threadIdx.x & 31;
    int n = lane & 15, half = lane >> 4;
    float bias = biasu[0];
#pragma unroll
    for (int it = 0; it < 8; ++it) {
        int idx = wave * 8 + it;        // 4 M-tiles(x) x 16 N-tiles(y)
        int mt = idx >> 4, nt = idx & 15;
        v16h a = load_a_frag(g1s + mt * 16 * 32, 32, lane);
        v16h bf = load_b_frag_nk(h2s + nt * 16 * 32, 32, lane);  // B[k=j][n=y]=h2s[y][j]
        v8f c = zero8();
        c = wmma16(a, bf, c);
#pragma unroll
        for (int r = 0; r < 8; ++r) {
            int x = xbase + mt * 16 + r + half * 8;
            int y = nt * 16 + n;
            __builtin_nontemporal_store(c[r] + bias, &out[((size_t)b * T_ + x) * T_ + y]);
        }
    }
}

// ---------------- Kernel 7: deprel biaffine (O=50), WMMA K=32 ----------------
// Per block (b,x): out[b,x,y,o] = h2r[b,y] . (Ur-contracted h1r[b,x])[.,o]
// Output chunk (256 y x 50 o) is fully contiguous -> coalesced, NT stores
// (420 MB write-once stream > 192 MB L2; keep Ur/h2 tiles resident instead).
__global__ void biaffine_r(const _Float16* __restrict__ h1r, const _Float16* __restrict__ h2r,
                           const float* __restrict__ Ur, const float* __restrict__ biasr,
                           float* __restrict__ out) {
    __shared__ __align__(16) _Float16 h2s[T_ * 32];       // 16 KB
    __shared__ __align__(16) _Float16 bm[32][66];         // B = U . h1 : (j x o), padded
    int tok = blockIdx.x;
    int b = tok >> 8;
    for (int i = threadIdx.x; i < T_ * 32 / 8; i += blockDim.x)
        async_copy_b128(&h2s[i * 8], &h2r[(size_t)b * T_ * 32 + i * 8]);
    const _Float16* h1row = h1r + (size_t)tok * 32;
    for (int e = threadIdx.x; e < 32 * 64; e += blockDim.x) {
        int j = e >> 6, o = e & 63;
        float acc = 0.f;
        if (o < O_ && j < 31) {
#pragma unroll 4
            for (int i2 = 0; i2 < 31; ++i2)
                acc += (float)h1row[i2] * Ur[(size_t)o * 961 + i2 * 31 + j];
        }
        bm[j][o] = (_Float16)acc;
    }
    async_wait0();
    __syncthreads();
    int wave = threadIdx.x >> 5, lane = threadIdx.x & 31;
    int n = lane & 15, half = lane >> 4;
#pragma unroll
    for (int it = 0; it < 8; ++it) {
        int idx = wave * 8 + it;            // 16 M-tiles(y) x 4 N-tiles(o)
        int mt = idx >> 2, nt = idx & 3;
        v16h a = load_a_frag(h2s + mt * 16 * 32, 32, lane);       // A[m=y][k=j]
        v16h bf = load_b_frag_kn(&bm[0][0] + nt * 16, 66, lane);  // B[k=j][n=o]
        v8f c = zero8();
        c = wmma16(a, bf, c);
        int o = nt * 16 + n;
        if (o < O_) {
            float bias = biasr[o];
#pragma unroll
            for (int r = 0; r < 8; ++r) {
                int y = mt * 16 + r + half * 8;
                __builtin_nontemporal_store(c[r] + bias,
                                            &out[((size_t)tok * T_ + y) * O_ + o]);
            }
        }
    }
}

// ---------------------------------- host ----------------------------------
extern "C" void kernel_launch(void* const* d_in, const int* in_sizes, int n_in,
                              void* d_out, int out_size, void* d_ws, size_t ws_size,
                              hipStream_t stream) {
    const float* hiddens = (const float*)d_in[0];
    const int*   b2t     = (const int*)  d_in[1];
    const float* W_ih_f  = (const float*)d_in[2];
    const float* W_hh_f  = (const float*)d_in[3];
    const float* b_ih_f  = (const float*)d_in[4];
    const float* b_hh_f  = (const float*)d_in[5];
    const float* W_ih_b  = (const float*)d_in[6];
    const float* W_hh_b  = (const float*)d_in[7];
    const float* b_ih_b  = (const float*)d_in[8];
    const float* b_hh_b  = (const float*)d_in[9];
    const float* W1u = (const float*)d_in[10]; const float* b1u = (const float*)d_in[11];
    const float* W2u = (const float*)d_in[12]; const float* b2u = (const float*)d_in[13];
    const float* Uu  = (const float*)d_in[14]; const float* biasu = (const float*)d_in[15];
    const float* W1r = (const float*)d_in[16]; const float* b1r = (const float*)d_in[17];
    const float* W2r = (const float*)d_in[18]; const float* b2r = (const float*)d_in[19];
    const float* Ur  = (const float*)d_in[20]; const float* biasr = (const float*)d_in[21];

    char* ws = (char*)d_ws;
    _Float16* bert_h = (_Float16*)ws;                         // 8192*768 f16 = 12,582,912 B
    _Float16* wih_h  = (_Float16*)(ws + 12582912);            // 2*400*768 f16 = 1,228,800 B
    float*    xg     = (float*)   (ws + 13811712);            // 2*8192*400 f32 = 26,214,400 B
    float*    h_all  = (float*)   (ws + 40026112);            // 2*8192*100 f32 = 6,553,600 B
    _Float16* h1u    = (_Float16*)(ws + 46579712);            // 4 x 8192*32 f16
    _Float16* h2u    = h1u + (size_t)NTOK * 32;
    _Float16* h1r    = h2u + (size_t)NTOK * 32;
    _Float16* h2r    = h1r + (size_t)NTOK * 32;

    float* out_u = (float*)d_out;                  // (B,T,T,1)
    float* out_r = out_u + (size_t)B_ * T_ * T_;   // (B,T,T,50)

    seg_mean_kernel<<<B_ * T_, 256, 0, stream>>>(hiddens, b2t, bert_h);
    convert_wih<<<(2 * G_ * D_ + 255) / 256, 256, 0, stream>>>(W_ih_f, W_ih_b, wih_h);
    xg_gemm<<<(2 * (NTOK / 16) * (G_ / 16)) / 8, 256, 0, stream>>>(
        bert_h, wih_h, b_ih_f, b_hh_f, b_ih_b, b_hh_b, xg);

    size_t recsm = (size_t)(G_ * WSTR + G_ + WSTR) * sizeof(float);   // ~168 KB of 320 KB WGP LDS
    hipFuncSetAttribute(reinterpret_cast<const void*>(lstm_rec),
                        hipFuncAttributeMaxDynamicSharedMemorySize, (int)recsm);
    lstm_rec<<<64, 512, recsm, stream>>>(W_hh_f, W_hh_b, xg, h_all);

    proj_kernel<<<B_ * T_, 128, 0, stream>>>(h_all, W1u, b1u, W2u, b2u, W1r, b1r, W2r, b2r,
                                             h1u, h2u, h1r, h2r);
    biaffine_u<<<B_ * 4, 256, 0, stream>>>(h1u, h2u, Uu, biasu, out_u);
    biaffine_r<<<B_ * T_, 256, 0, stream>>>(h1r, h2r, Ur, biasr, out_r);
}
